// Decoder_11811160064753
// MI455X (gfx1250) — matrix operand order repaired and checked by reading, not compiled, not run on hardware
//
#include <hip/hip_runtime.h>
#include <hip/hip_bf16.h>

// ---------------------------------------------------------------------------
// MI455X (gfx1250, CDNA5) decoder forward.
// All convolutions -> implicit GEMM on V_WMMA_F32_16X16X32_F16 (wave32),
// f16 inputs / f32 accumulate.
// Data layout engineered for the WMMA fragment shapes:
//   * activations: padded NHWC f16  [B, Hp, Wp, C]
//   * weights    : tap-major f16    [Ot*16, khw*C]   (k = tap*C + c)
// All C and K in this net are multiples of 32 => every lane's B fragment is a
// single aligned 32B contiguous load, every A fragment two aligned 16B loads.
// ---------------------------------------------------------------------------

typedef __attribute__((ext_vector_type(16))) _Float16 v16h;
typedef __attribute__((ext_vector_type(8)))  _Float16 v8h;
typedef __attribute__((ext_vector_type(8)))  float    v8f;

#define EPS_INORM 1e-5f

// ---------------------------------------------------------------------------
// weight transform: src f32 [O, Cdim, khw]  ->  dst f16 [OP, khw*Cdim]
// dst[o, t*Cdim + c] = src[o, c, t];  rows o >= O are zero-filled (OP = Ot*16)
// (khw==1 makes this a plain convert, used for the deformable 1x1 GEMMs.)
// ---------------------------------------------------------------------------
__global__ __launch_bounds__(256)
void wgt_f16_kernel(const float* __restrict__ src, _Float16* __restrict__ dst,
                    int O, int OP, int Cdim, int khw) {
  long long K = (long long)Cdim * khw;
  long long total = (long long)OP * K;
  long long idx = (long long)blockIdx.x * blockDim.x + threadIdx.x;
  if (idx >= total) return;
  long long kd = idx % K;
  int o = (int)(idx / K);
  int t = (int)(kd / Cdim);
  int c = (int)(kd - (long long)t * Cdim);
  float v = 0.f;
  if (o < O) v = src[((long long)o * Cdim + c) * khw + t];
  dst[idx] = (_Float16)v;
}

// ---------------------------------------------------------------------------
// pad (reflect or constant-zero) + f32->f16 + NCHW -> NHWC
// src [B,C,H,W] f32 -> dst [B,Hp,Wp,C] f16
// ---------------------------------------------------------------------------
__global__ __launch_bounds__(256)
void pad_nhwc_f16_kernel(const float* __restrict__ src, _Float16* __restrict__ dst,
                         int B, int C, int H, int W, int p, int reflect) {
  int Hp = H + 2 * p, Wp = W + 2 * p;
  long long total = (long long)B * Hp * Wp * C;
  long long idx = (long long)blockIdx.x * blockDim.x + threadIdx.x;
  if (idx >= total) return;
  int c = (int)(idx % C);
  long long t = idx / C;
  int x = (int)(t % Wp);
  t /= Wp;
  int y = (int)(t % Hp);
  int b = (int)(t / Hp);
  int yy = y - p, xx = x - p;
  float v = 0.0f;
  if (reflect) {
    if (yy < 0) yy = -yy;
    if (yy >= H) yy = 2 * H - 2 - yy;
    if (xx < 0) xx = -xx;
    if (xx >= W) xx = 2 * W - 2 - xx;
    v = src[(((long long)b * C + c) * H + yy) * W + xx];
  } else {
    if (yy >= 0 && yy < H && xx >= 0 && xx < W)
      v = src[(((long long)b * C + c) * H + yy) * W + xx];
  }
  dst[idx] = (_Float16)v;
}

// ---------------------------------------------------------------------------
// Implicit-GEMM conv via WMMA.
//   in   : f16 NHWC padded [B, Hp, Wp, C]
//   wgt  : f16 [Ot*16, khw*C], tap-major K, rows >= O zeroed
//   bias : f32 [O] ; out : f32 NCHW [B, O, H, W]
// One wave -> 16(O) x 64(N) tile; A fragment reused across 4 WMMAs.
// Fragment layouts per CDNA5 ISA (wave32):
//   A 16x32 f16 : lane L -> M=L%16 ; hw-K = base(0|8) + {e, e+8}
//   B 32x16 f16 : lane L -> N=L%16 ; K used here for A and B = base(0|16) + e (contiguous)
//   C/D f32     : VGPR v -> M = v (lanes 0-15) / v+8 (lanes 16-31), N=L%16
// Requires: C % 32 == 0 (holds for every conv in this net). act: 1 = tanh.
// ---------------------------------------------------------------------------
__global__ __launch_bounds__(256)
void conv_wmma_kernel(const _Float16* __restrict__ in,
                      const _Float16* __restrict__ wgt,
                      const float* __restrict__ bias,
                      float* __restrict__ out,
                      int B, int C, int Hp, int Wp, int H, int W,
                      int O, int kh, int kw, int act) {
  const int lane   = threadIdx.x & 31;
  const int waveId = blockIdx.x * (blockDim.x >> 5) + (threadIdx.x >> 5);
  const int Ot = (O + 15) >> 4;
  const int N  = B * H * W;
  if (waveId >= Ot * (N >> 6)) return;
  const int ot = waveId % Ot;
  const int nt = waveId / Ot;

  const int khw = kh * kw;
  const int K   = khw * C;          // multiple of 32
  const int HW  = H * W;
  const int ln16 = lane & 15;
  const int kbaseA = (lane < 16) ? 0 : 16;
  const int kbaseB = (lane < 16) ? 0 : 16;
  const _Float16* wrow = wgt + (long long)(ot * 16 + ln16) * K + kbaseA;

  // pixel coords of the 4 N-subtiles owned by this lane (N = lane%16)
  int pb[4], py_[4], px_[4];
  const _Float16* pixp[4];
  #pragma unroll
  for (int j = 0; j < 4; ++j) {
    int n = (nt << 6) + (j << 4) + ln16;
    px_[j] = n % W;
    int t  = n / W;
    py_[j] = t % H;
    pb[j]  = t / H;
    pixp[j] = in + (((long long)pb[j] * Hp + py_[j]) * Wp + px_[j]) * C;
  }

  v8f zero8 = {0.f, 0.f, 0.f, 0.f, 0.f, 0.f, 0.f, 0.f};
  v8f acc[4] = {zero8, zero8, zero8, zero8};

  // incremental tap/channel tracking: k0 = kk + kbaseB, tap = k0/C, c0 = k0%C
  int c0 = kbaseB, kx = 0, ky = 0;
  for (int kk = 0; kk < K; kk += 32) {
    if (kk + 32 < K) __builtin_prefetch(wrow + kk + 32, 0, 1);

    // A fragment: two contiguous aligned 16B loads ({0..15} contiguous pattern)
    v8h a0 = *(const v8h*)(wrow + kk);
    v8h a1 = *(const v8h*)(wrow + kk + 8);
    v16h a = __builtin_shufflevector(a0, a1, 0, 1, 2, 3, 4, 5, 6, 7,
                                             8, 9, 10, 11, 12, 13, 14, 15);

    // B fragments: one contiguous aligned 32B load per N-subtile
    const int boff = (ky * Wp + kx) * C + c0;
    #pragma unroll
    for (int j = 0; j < 4; ++j) {
      v16h bm = *(const v16h*)(pixp[j] + boff);
      acc[j] = __builtin_amdgcn_wmma_f32_16x16x32_f16(
          false, a, false, bm, (short)0, acc[j], false, false);
    }

    c0 += 32;
    if (c0 >= C) { c0 -= C; ++kx; if (kx == kw) { kx = 0; ++ky; } }
  }

  // epilogue: bias (+ tanh), scatter to NCHW f32
  #pragma unroll
  for (int j = 0; j < 4; ++j) {
    long long ob = ((long long)pb[j] * O) * HW + (long long)py_[j] * W + px_[j];
    #pragma unroll
    for (int v = 0; v < 8; ++v) {
      int oo = ot * 16 + ((lane < 16) ? v : (v + 8));
      if (oo < O) {
        float r = acc[j][v] + bias[oo];
        if (act) r = tanhf(r);
        out[ob + (long long)oo * HW] = r;
      }
    }
  }
}

// ---------------------------------------------------------------------------
// InstanceNorm2d (no affine), optional ReLU. One block per (b,c). NCHW f32.
// ---------------------------------------------------------------------------
__global__ __launch_bounds__(256)
void inorm_kernel(const float* __restrict__ x, float* __restrict__ y,
                  int HW, int relu) {
  __shared__ float ss[256], ss2[256];
  long long base = (long long)blockIdx.x * HW;
  float s = 0.f, s2 = 0.f;
  for (int i = threadIdx.x; i < HW; i += 256) {
    float v = x[base + i];
    s += v; s2 += v * v;
  }
  ss[threadIdx.x] = s; ss2[threadIdx.x] = s2;
  __syncthreads();
  for (int st = 128; st > 0; st >>= 1) {
    if (threadIdx.x < st) { ss[threadIdx.x] += ss[threadIdx.x + st];
                            ss2[threadIdx.x] += ss2[threadIdx.x + st]; }
    __syncthreads();
  }
  float m   = ss[0] / (float)HW;
  float var = ss2[0] / (float)HW - m * m;
  float inv = rsqrtf(fmaxf(var, 0.f) + EPS_INORM);
  for (int i = threadIdx.x; i < HW; i += 256) {
    float v = (x[base + i] - m) * inv;
    if (relu) v = fmaxf(v, 0.f);
    y[base + i] = v;
  }
}

__global__ __launch_bounds__(256)
void add_inplace_kernel(float* __restrict__ dst, const float* __restrict__ src,
                        long long n) {
  long long i = (long long)blockIdx.x * blockDim.x + threadIdx.x;
  if (i < n) dst[i] += src[i];
}

// nearest upsample x2 : in [B,C,H,W] -> out [B,C,2H,2W]  (NCHW f32)
__global__ __launch_bounds__(256)
void upsample2_kernel(const float* __restrict__ in, float* __restrict__ out,
                      int BC, int H, int W) {
  int W2 = 2 * W, H2 = 2 * H;
  long long total = (long long)BC * H2 * W2;
  long long idx = (long long)blockIdx.x * blockDim.x + threadIdx.x;
  if (idx >= total) return;
  int x = (int)(idx % W2);
  int y = (int)((idx / W2) % H2);
  long long bc = idx / ((long long)W2 * H2);
  out[idx] = in[bc * H * W + (long long)(y >> 1) * W + (x >> 1)];
}

// channel concat: out[:,0:C1]=a, out[:,C1:C1+C2]=b  (NCHW f32)
__global__ __launch_bounds__(256)
void concat_kernel(const float* __restrict__ a, const float* __restrict__ b,
                   float* __restrict__ out, int B, int C1, int C2, int HW) {
  int Co = C1 + C2;
  long long total = (long long)B * Co * HW;
  long long idx = (long long)blockIdx.x * blockDim.x + threadIdx.x;
  if (idx >= total) return;
  int pix = (int)(idx % HW);
  int ch  = (int)((idx / HW) % Co);
  long long bb = idx / ((long long)HW * Co);
  out[idx] = (ch < C1) ? a[(bb * C1 + ch) * HW + pix]
                       : b[(bb * C2 + (ch - C1)) * HW + pix];
}

// ---------------------------------------------------------------------------
// Deformable-conv sampling -> masked bilinear "col" tensor in NHWC(C*9) f16:
//   col[((b*H + y)*W + x)*C*9 + (c*9 + k)]   (matches W[O,C,9] identity flatten,
//   consumed by conv_wmma_kernel as a 1x1 conv with C' = C*9)
// One thread per (b,k,y,x); loops over C, 4 taps computed once.
// ---------------------------------------------------------------------------
__global__ __launch_bounds__(256)
void dcn_col_kernel(const float* __restrict__ skip, const float* __restrict__ om,
                    _Float16* __restrict__ col, int B, int C, int H, int W) {
  int HW = H * W;
  long long total = (long long)B * 9 * HW;
  long long idx = (long long)blockIdx.x * blockDim.x + threadIdx.x;
  if (idx >= total) return;
  int x = (int)(idx % W);
  int y = (int)((idx / W) % H);
  int k = (int)((idx / HW) % 9);
  int b = (int)(idx / ((long long)HW * 9));

  const float* omb = om + (long long)b * 27 * HW;
  int p = y * W + x;
  float dy = omb[(2 * k) * HW + p];
  float dx = omb[(2 * k + 1) * HW + p];
  float mk = omb[(18 + k) * HW + p];
  mk = 1.0f / (1.0f + expf(-mk));

  float py = (float)y + (float)(k / 3 - 1) + dy;
  float px = (float)x + (float)(k % 3 - 1) + dx;
  float y0f = floorf(py), x0f = floorf(px);
  int y0 = (int)y0f, x0 = (int)x0f;
  float wy1 = py - y0f, wx1 = px - x0f;
  float wy0 = 1.f - wy1, wx0 = 1.f - wx1;

  bool vy0 = (y0 >= 0) && (y0 < H);
  bool vy1 = (y0 + 1 >= 0) && (y0 + 1 < H);
  bool vx0 = (x0 >= 0) && (x0 < W);
  bool vx1 = (x0 + 1 >= 0) && (x0 + 1 < W);
  int y0c = min(max(y0, 0), H - 1);
  int y1c = min(max(y0 + 1, 0), H - 1);
  int x0c = min(max(x0, 0), W - 1);
  int x1c = min(max(x0 + 1, 0), W - 1);
  float w00 = wy0 * wx0 * ((vy0 && vx0) ? 1.f : 0.f);
  float w01 = wy0 * wx1 * ((vy0 && vx1) ? 1.f : 0.f);
  float w10 = wy1 * wx0 * ((vy1 && vx0) ? 1.f : 0.f);
  float w11 = wy1 * wx1 * ((vy1 && vx1) ? 1.f : 0.f);
  int o00 = y0c * W + x0c, o01 = y0c * W + x1c;
  int o10 = y1c * W + x0c, o11 = y1c * W + x1c;

  const float* sb = skip + (long long)b * C * HW;
  _Float16* cp = col + ((long long)p + (long long)b * HW) * (C * 9) + k;
  for (int c = 0; c < C; ++c) {
    const float* s = sb + (long long)c * HW;
    float g = w00 * s[o00] + w01 * s[o01] + w10 * s[o10] + w11 * s[o11];
    cp[c * 9] = (_Float16)(g * mk);
  }
}

// ---------------------------------------------------------------------------
// mean(|offset|) over channels 0..17 of a [B,27,H,W] tensor, scaled, atomic-add
// ---------------------------------------------------------------------------
__global__ __launch_bounds__(256)
void absmean_add_kernel(const float* __restrict__ om, int B, int HW,
                        float scale, float* __restrict__ acc) {
  __shared__ float ss[256];
  long long total = (long long)B * 18 * HW;
  float s = 0.f;
  for (long long i = (long long)blockIdx.x * 256 + threadIdx.x; i < total;
       i += (long long)gridDim.x * 256) {
    long long b  = i / (18LL * HW);
    long long r  = i - b * 18LL * HW;
    long long ch = r / HW;
    long long px = r - ch * HW;
    s += fabsf(om[(b * 27 + ch) * HW + px]);
  }
  ss[threadIdx.x] = s;
  __syncthreads();
  for (int st = 128; st > 0; st >>= 1) {
    if (threadIdx.x < st) ss[threadIdx.x] += ss[threadIdx.x + st];
    __syncthreads();
  }
  if (threadIdx.x == 0) atomicAdd(acc, ss[0] * scale);
}

__global__ void zero1_kernel(float* p) { p[0] = 0.f; }

// ---------------------------------------------------------------------------
// Host sequencing
// ---------------------------------------------------------------------------
extern "C" void kernel_launch(void* const* d_in, const int* in_sizes, int n_in,
                              void* d_out, int out_size, void* d_ws, size_t ws_size,
                              hipStream_t stream) {
  (void)in_sizes; (void)n_in; (void)out_size;
  const int B = 8;

  const float* x     = (const float*)d_in[0];
  const float* skip1 = (const float*)d_in[1];
  const float* skip2 = (const float*)d_in[2];
  const float* rb_w1[2] = {(const float*)d_in[3], (const float*)d_in[7]};
  const float* rb_b1[2] = {(const float*)d_in[4], (const float*)d_in[8]};
  const float* rb_w2[2] = {(const float*)d_in[5], (const float*)d_in[9]};
  const float* rb_b2[2] = {(const float*)d_in[6], (const float*)d_in[10]};
  const float* conv1_w = (const float*)d_in[11]; const float* conv1_b = (const float*)d_in[12];
  const float* conv3_w = (const float*)d_in[13]; const float* conv3_b = (const float*)d_in[14];
  const float* conv5_w = (const float*)d_in[15]; const float* conv5_b = (const float*)d_in[16];
  const float* conv6_w = (const float*)d_in[17]; const float* conv6_b = (const float*)d_in[18];
  const float* dcn2_off_w = (const float*)d_in[19]; const float* dcn2_off_b = (const float*)d_in[20];
  const float* dcn2_w = (const float*)d_in[21]; const float* dcn2_b = (const float*)d_in[22];
  const float* dcn1_off_w = (const float*)d_in[23]; const float* dcn1_off_b = (const float*)d_in[24];
  const float* dcn1_w = (const float*)d_in[25]; const float* dcn1_b = (const float*)d_in[26];

  // ---- workspace bump allocator ----
  char* base = (char*)d_ws;
  size_t cur = 0;
  auto alloc = [&](size_t bytes) -> void* {
    void* p = base + cur;
    cur += (bytes + 255) & ~(size_t)255;
    return p;
  };

  // f16 weights, tap-major, rows padded to Ot*16 and zero-filled
  _Float16* h_rbw1[2]; _Float16* h_rbw2[2];
  for (int r = 0; r < 2; ++r) {
    h_rbw1[r] = (_Float16*)alloc((size_t)256 * 2304 * 2);
    h_rbw2[r] = (_Float16*)alloc((size_t)256 * 2304 * 2);
  }
  _Float16* h_c1w = (_Float16*)alloc((size_t)256 * 6400 * 2);
  _Float16* h_c3w = (_Float16*)alloc((size_t)128 * 12800 * 2);
  _Float16* h_c5w = (_Float16*)alloc((size_t)64 * 6400 * 2);
  _Float16* h_c6w = (_Float16*)alloc((size_t)16 * 3136 * 2);   // O=3 -> 16 rows
  _Float16* h_o2w = (_Float16*)alloc((size_t)32 * 4608 * 2);   // O=27 -> 32 rows
  _Float16* h_d2w = (_Float16*)alloc((size_t)256 * 2304 * 2);
  _Float16* h_o1w = (_Float16*)alloc((size_t)32 * 2304 * 2);   // O=27 -> 32 rows
  _Float16* h_d1w = (_Float16*)alloc((size_t)128 * 1152 * 2);

  float* om2 = (float*)alloc((size_t)B * 27 * 32 * 32 * 4);    // kept until reduction
  float* om1 = (float*)alloc((size_t)B * 27 * 64 * 64 * 4);

  // reusable arenas (lifetimes annotated at use sites)
  char* M1 = (char*)alloc(134217728);  // up256@128
  char* M2 = (char*)alloc( 75497472);  // col2 -> col1 -> pad_c5
  char* M3 = (char*)alloc( 67108864);  // up512@64 -> c5norm
  char* M4 = (char*)alloc( 37879808);  // all pad buffers
  char* M5 = (char*)alloc( 33554432);  // cat256@64 -> cat1 -> c5out
  char* M6 = (char*)alloc( 16777216);  // conv tmp@32 -> cat512@32 -> cat2 -> cur64
  char* M7 = (char*)alloc( 16777216);  // inorm tmp -> pre2 -> c3out -> pre1
  char* M8 = (char*)alloc(  8388608);  // cur32 (256ch@32)

  if (cur > ws_size) return;  // workspace too small to compute anything

  auto wcvt = [&](const float* s, _Float16* d, int O, int OP, int Cdim, int khw) {
    long long n = (long long)OP * Cdim * khw;
    wgt_f16_kernel<<<dim3((unsigned)((n + 255) / 256)), dim3(256), 0, stream>>>(
        s, d, O, OP, Cdim, khw);
  };
  auto pad = [&](const float* s, _Float16* d, int C, int H, int W, int p, int refl) {
    long long n = (long long)B * (H + 2 * p) * (W + 2 * p) * C;
    pad_nhwc_f16_kernel<<<dim3((unsigned)((n + 255) / 256)), dim3(256), 0, stream>>>(
        s, d, B, C, H, W, p, refl);
  };
  auto conv = [&](const _Float16* in, const _Float16* w, const float* bias, float* out,
                  int C, int Hp, int Wp, int H, int W, int O, int kh, int kw, int act) {
    int Ot = (O + 15) / 16;
    int waves = Ot * ((B * H * W) / 64);
    conv_wmma_kernel<<<dim3((waves + 7) / 8), dim3(256), 0, stream>>>(
        in, w, bias, out, B, C, Hp, Wp, H, W, O, kh, kw, act);
  };
  auto inorm = [&](const float* xx, float* yy, int C, int HW, int relu) {
    inorm_kernel<<<dim3(B * C), dim3(256), 0, stream>>>(xx, yy, HW, relu);
  };
  auto ew = [&](long long n) { return dim3((unsigned)((n + 255) / 256)); };

  // ---- weight conversion (tap-major transpose + f16 + row padding) ----
  for (int r = 0; r < 2; ++r) {
    wcvt(rb_w1[r], h_rbw1[r], 256, 256, 256, 9);
    wcvt(rb_w2[r], h_rbw2[r], 256, 256, 256, 9);
  }
  wcvt(conv1_w, h_c1w, 256, 256, 256, 25);
  wcvt(conv3_w, h_c3w, 128, 128, 512, 25);
  wcvt(conv5_w, h_c5w, 64, 64, 256, 25);
  wcvt(conv6_w, h_c6w, 3, 16, 64, 49);
  wcvt(dcn2_off_w, h_o2w, 27, 32, 512, 9);
  wcvt(dcn1_off_w, h_o1w, 27, 32, 256, 9);
  wcvt(dcn2_w, h_d2w, 256, 256, 2304, 1);   // identity flatten [O, C*9]
  wcvt(dcn1_w, h_d1w, 128, 128, 1152, 1);

  // ---- stage 1: residual blocks (256ch @32x32) ----
  float* cur32 = (float*)M8;
  hipMemcpyAsync(cur32, x, (size_t)B * 256 * 32 * 32 * 4, hipMemcpyDeviceToDevice, stream);
  _Float16* padb = (_Float16*)M4;
  float* convt = (float*)M6;
  float* ntmp  = (float*)M7;
  for (int r = 0; r < 2; ++r) {
    pad(cur32, padb, 256, 32, 32, 1, 1);
    conv(padb, h_rbw1[r], rb_b1[r], convt, 256, 34, 34, 32, 32, 256, 3, 3, 0);
    inorm(convt, ntmp, 256, 1024, 1);                    // h = relu(inorm(conv1))
    pad(ntmp, padb, 256, 32, 32, 1, 1);
    conv(padb, h_rbw2[r], rb_b2[r], convt, 256, 34, 34, 32, 32, 256, 3, 3, 0);
    inorm(convt, ntmp, 256, 1024, 0);
    long long n = (long long)B * 256 * 1024;
    add_inplace_kernel<<<ew(n), dim3(256), 0, stream>>>(cur32, ntmp, n);
  }

  // ---- stage 2: conv1 (5x5, 256->256 @32) + inorm + relu ----
  pad(cur32, padb, 256, 32, 32, 2, 1);
  conv(padb, h_c1w, conv1_b, convt, 256, 36, 36, 32, 32, 256, 5, 5, 0);
  inorm(convt, cur32, 256, 1024, 1);

  // ---- stage 3: dcn_2 block ----
  float* cat512 = (float*)M6;          // convt dead
  concat_kernel<<<ew((long long)B * 512 * 1024), dim3(256), 0, stream>>>(
      cur32, skip2, cat512, B, 256, 256, 1024);
  pad(cat512, padb, 512, 32, 32, 1, 0);                 // constant pad
  conv(padb, h_o2w, dcn2_off_b, om2, 512, 34, 34, 32, 32, 27, 3, 3, 0);
  _Float16* col2 = (_Float16*)M2;
  dcn_col_kernel<<<ew((long long)B * 9 * 1024), dim3(256), 0, stream>>>(
      skip2, om2, col2, B, 256, 32, 32);
  float* pre2 = (float*)M7;
  conv(col2, h_d2w, dcn2_b, pre2, 2304, 32, 32, 32, 32, 256, 1, 1, 0);  // 1x1 GEMM
  float* cat2 = (float*)M6;            // cat512 dead (om2 computed)
  concat_kernel<<<ew((long long)B * 512 * 1024), dim3(256), 0, stream>>>(
      pre2, cur32, cat2, B, 256, 256, 1024);

  // ---- stage 4: upsample x2 -> 512ch @64 ----
  float* up64 = (float*)M3;
  upsample2_kernel<<<ew((long long)B * 512 * 4096), dim3(256), 0, stream>>>(
      cat2, up64, B * 512, 32, 32);

  // ---- stage 5: conv3 (5x5, 512->128 @64) + inorm + relu ----
  pad(up64, padb, 512, 64, 64, 2, 1);
  float* c3out = (float*)M7;
  conv(padb, h_c3w, conv3_b, c3out, 512, 68, 68, 64, 64, 128, 5, 5, 0);
  float* cur64 = (float*)M6;           // cat2 dead
  inorm(c3out, cur64, 128, 4096, 1);

  // ---- stage 6: dcn_1 block ----
  float* cat256 = (float*)M5;
  concat_kernel<<<ew((long long)B * 256 * 4096), dim3(256), 0, stream>>>(
      cur64, skip1, cat256, B, 128, 128, 4096);
  pad(cat256, padb, 256, 64, 64, 1, 0);
  conv(padb, h_o1w, dcn1_off_b, om1, 256, 66, 66, 64, 64, 27, 3, 3, 0);
  _Float16* col1 = (_Float16*)M2;      // col2 dead
  dcn_col_kernel<<<ew((long long)B * 9 * 4096), dim3(256), 0, stream>>>(
      skip1, om1, col1, B, 128, 64, 64);
  float* pre1 = (float*)M7;            // c3out dead
  conv(col1, h_d1w, dcn1_b, pre1, 1152, 64, 64, 64, 64, 128, 1, 1, 0);
  float* cat1 = (float*)M5;            // cat256 dead (om1 computed)
  concat_kernel<<<ew((long long)B * 256 * 4096), dim3(256), 0, stream>>>(
      pre1, cur64, cat1, B, 128, 128, 4096);

  // ---- stage 7: upsample x2 -> 256ch @128 ----
  float* up128 = (float*)M1;
  upsample2_kernel<<<ew((long long)B * 256 * 16384), dim3(256), 0, stream>>>(
      cat1, up128, B * 256, 64, 64);

  // ---- stage 8: conv5 (5x5, 256->64 @128) + inorm + relu ----
  _Float16* padc5 = (_Float16*)M2;     // col1 dead
  pad(up128, padc5, 256, 128, 128, 2, 1);
  float* c5out = (float*)M5;           // cat1 dead
  conv(padc5, h_c5w, conv5_b, c5out, 256, 132, 132, 128, 128, 64, 5, 5, 0);
  float* c5n = (float*)M3;             // up64 dead
  inorm(c5out, c5n, 64, 16384, 1);

  // ---- stage 9: conv6 (7x7, 64->3 @128) + tanh -> d_out ----
  pad(c5n, padb, 64, 128, 128, 3, 1);
  conv(padb, h_c6w, conv6_b, (float*)d_out, 64, 134, 134, 128, 128, 3, 7, 7, 1);

  // ---- offset_sum = 0.5*(mean|offset1| + mean|offset2|) -> d_out[393216] ----
  float* acc = (float*)d_out + (size_t)B * 3 * 128 * 128;
  zero1_kernel<<<1, 1, 0, stream>>>(acc);
  absmean_add_kernel<<<dim3(1024), dim3(256), 0, stream>>>(
      om2, B, 1024, 0.5f / (float)((long long)B * 18 * 1024), acc);
  absmean_add_kernel<<<dim3(1024), dim3(256), 0, stream>>>(
      om1, B, 4096, 0.5f / (float)((long long)B * 18 * 4096), acc);
}
